// RelativeMultiHeadAttn_17386027614584
// MI455X (gfx1250) — matrix operand order repaired and checked by reading, not compiled, not run on hardware
//
#include <hip/hip_runtime.h>

typedef __bf16 bf16_t;
typedef __attribute__((ext_vector_type(16))) __bf16 v16bf;
typedef __attribute__((ext_vector_type(8)))  __bf16 v8bf;
typedef __attribute__((ext_vector_type(8)))  float  v8f;

#define B_SZ 2
#define L_SZ 1024
#define H_SZ 16
#define DH_SZ 64
#define D_SZ 1024
#define TWO_L 2048

__device__ __forceinline__ v8f zero8() {
    v8f z;
    for (int i = 0; i < 8; ++i) z[i] = 0.0f;
    return z;
}

__device__ __forceinline__ v8f wmma_bf16(v16bf a, v16bf b, v8f c) {
    // D = A(16x32 bf16) x B(32x16 bf16) + C(16x16 f32)
    return __builtin_amdgcn_wmma_f32_16x16x32_bf16(
        false, a, false, b, (short)0, c, false, false);
}

// A-matrix fragment (16x32 bf16, M x K), row-major source, `row` = absolute row.
// Lane layout: lane<16 holds K 0..15; lane>=16 holds K 16..31.
__device__ __forceinline__ v16bf load_a_frag(const bf16_t* base, int row, int ld) {
    int lane = threadIdx.x & 31;
    const bf16_t* p = base + (size_t)row * ld + ((lane >> 4) << 4);
    return *(const v16bf*)p;
}

// B-matrix fragment (32x16 bf16, K x N) from N-major storage (row n holds K contiguous).
// Lane layout: lanes 0-15 -> col n, K 0..15; lanes 16-31 -> col n, K 16..31.
__device__ __forceinline__ v16bf load_b_frag(const bf16_t* base, int ld) {
    int lane = threadIdx.x & 31;
    const bf16_t* p = base + (size_t)(lane & 15) * ld + ((lane >> 4) << 4);
    return *(const v16bf*)p;
}

__global__ void convert_kernel(const float* __restrict__ src, bf16_t* __restrict__ dst, int n) {
    int i = blockIdx.x * blockDim.x + threadIdx.x;
    if (i < n) dst[i] = (bf16_t)src[i];
}

// pos_embed[l][d], l in [0,2L): pval = l - L. cols 0..31 sin, 32..63 cos.
// Dmat[h][l] = r_w_bias[h] . pos_embed[l]
__global__ void pe_kernel(const float* __restrict__ rwb,
                          bf16_t* __restrict__ pebf, float* __restrict__ Dmat) {
    int l = blockIdx.x * blockDim.x + threadIdx.x;
    if (l >= TWO_L) return;
    float pv = (float)(l - L_SZ);
    float row[64];
    for (int i = 0; i < 32; ++i) {
        float f = __expf(-0.29710775393471563f * (float)i); // -ln(10000)/31
        float a = pv * f;
        row[i]      = __sinf(a);
        row[i + 32] = __cosf(a);
    }
    for (int d = 0; d < 64; ++d) pebf[(size_t)l * 64 + d] = (bf16_t)row[d];
    for (int h = 0; h < H_SZ; ++h) {
        float s = 0.f;
        for (int d = 0; d < 64; ++d) s += rwb[h * 64 + d] * row[d];
        Dmat[(size_t)h * TWO_L + l] = s;
    }
}

// qv = x @ Wqv^T.  M = B*L = 2048, N = 2048, K = 1024.
// grid (M/16, N/256), block 128 (4 waves). Each wave computes a 16(m) x 64(n)
// strip with 4 accumulators, reusing one A fragment across 4 B fragments per
// k-step (4 independent WMMA chains -> hides XDL latency, 4x fewer A loads).
__global__ void __launch_bounds__(128)
qv_gemm_kernel(const bf16_t* __restrict__ xbf, const bf16_t* __restrict__ wbf,
               const float* __restrict__ rrb,
               bf16_t* __restrict__ qbf, bf16_t* __restrict__ qprbf,
               bf16_t* __restrict__ vtbf) {
    int lane = threadIdx.x & 31;
    int wave = threadIdx.x >> 5;
    int m0 = blockIdx.x * 16;
    int n0 = blockIdx.y * 256 + wave * 64;
    const bf16_t* abase = xbf + (size_t)m0 * D_SZ;
    const bf16_t* bbase = wbf + (size_t)n0 * D_SZ;
    v8f acc0 = zero8(), acc1 = zero8(), acc2 = zero8(), acc3 = zero8();
    for (int k0 = 0; k0 < D_SZ; k0 += 32) {
        v16bf a  = load_a_frag(abase + k0, lane & 15, D_SZ);
        v16bf b0 = load_b_frag(bbase + k0,                   D_SZ);
        v16bf b1 = load_b_frag(bbase + (size_t)16 * D_SZ + k0, D_SZ);
        v16bf b2 = load_b_frag(bbase + (size_t)32 * D_SZ + k0, D_SZ);
        v16bf b3 = load_b_frag(bbase + (size_t)48 * D_SZ + k0, D_SZ);
        acc0 = wmma_bf16(a, b0, acc0);
        acc1 = wmma_bf16(a, b1, acc1);
        acc2 = wmma_bf16(a, b2, acc2);
        acc3 = wmma_bf16(a, b3, acc3);
    }
    int mbase = m0 + (lane >> 4) * 8;
    for (int sub = 0; sub < 4; ++sub) {
        v8f acc = (sub == 0) ? acc0 : (sub == 1) ? acc1 : (sub == 2) ? acc2 : acc3;
        int n = n0 + sub * 16 + (lane & 15);
        for (int j = 0; j < 8; ++j) {
            int m = mbase + j;
            int bidx = m >> 10, l = m & 1023;
            float val = acc[j];
            if (n < D_SZ) { // q half
                int hh = n >> 6, d = n & 63;
                size_t idx = ((((size_t)bidx * H_SZ + hh) * L_SZ) + l) * DH_SZ + d;
                qbf[idx]   = (bf16_t)val;
                qprbf[idx] = (bf16_t)(val + rrb[hh * 64 + d]);
            } else {        // v half, stored transposed: (b,h,dh,L)
                int nn = n - D_SZ;
                int hh = nn >> 6, d = nn & 63;
                vtbf[(((size_t)bidx * H_SZ + hh) * DH_SZ + d) * L_SZ + l] = (bf16_t)val;
            }
        }
    }
}

// One workgroup per (q-tile, h, b). 4 waves.
__global__ void __launch_bounds__(128)
attn_kernel(const bf16_t* __restrict__ xbf,   // K source (b, l, 1024)
            const bf16_t* __restrict__ qbf,   // (b,h,L,64)
            const bf16_t* __restrict__ qprbf, // q + r_r_bias
            const bf16_t* __restrict__ vtbf,  // (b,h,64,L)
            const bf16_t* __restrict__ pebf,  // (2L,64)
            const float*  __restrict__ Dmat,  // (H,2L)
            const int*    __restrict__ mask,  // (B,L)
            float* __restrict__ out)          // (B,L,1024)
{
    __shared__ __align__(32) float  s_score[16][L_SZ]; // 64 KB
    __shared__ __align__(32) bf16_t s_p[16][L_SZ];     // 32 KB
    __shared__ float s_red[16][16];

    int qt = blockIdx.x, h = blockIdx.y, b = blockIdx.z;
    int q0 = qt * 16;
    int tid = threadIdx.x;
    int wave = tid >> 5;
    int lane = tid & 31;

    const bf16_t* qmat   = qbf   + ((size_t)(b * H_SZ + h) * L_SZ) * DH_SZ;
    const bf16_t* qprmat = qprbf + ((size_t)(b * H_SZ + h) * L_SZ) * DH_SZ;

    // A fragments (K = 64 -> two 16x32 chunks each)
    int rA = q0 + (lane & 15);
    v16bf aAC0 = load_a_frag(qprmat, rA, DH_SZ);
    v16bf aAC1 = load_a_frag(qprmat + 32, rA, DH_SZ);
    v16bf aS0  = load_a_frag(qmat, rA, DH_SZ);
    v16bf aS1  = load_a_frag(qmat + 32, rA, DH_SZ);
    int rB = q0 - 1 + (lane & 15);
    if (rB < 0) rB = 0; // q0==0, row unused (no valid k for m=0)
    v16bf aB0 = load_a_frag(qmat, rB, DH_SZ);
    v16bf aB1 = load_a_frag(qmat + 32, rB, DH_SZ);

    // ---- Phase 1: AC = (q + r_r_bias) @ k^T ----
    for (int kt = wave; kt < 64; kt += 4) {
        int k0 = kt * 16;
        const bf16_t* kbase = xbf + ((size_t)b * L_SZ + k0) * D_SZ + h * 64;
        v16bf b0 = load_b_frag(kbase, D_SZ);
        v16bf b1 = load_b_frag(kbase + 32, D_SZ);
        v8f acc = zero8();
        acc = wmma_bf16(aAC0, b0, acc);
        acc = wmma_bf16(aAC1, b1, acc);
        int n = lane & 15, mb = (lane >> 4) * 8;
        for (int j = 0; j < 8; ++j) s_score[mb + j][k0 + n] = acc[j];
    }
    __syncthreads();

    // ---- Phase 2: main shift band: score[m][l+q0+m] += q[q].pe[l] + D[h][l] ----
    int ntile2 = 64 - qt; // l up to L-1-q0
    for (int lt = wave; lt < ntile2; lt += 4) {
        int l0 = lt * 16;
        const bf16_t* pbase = pebf + (size_t)l0 * 64;
        v16bf b0 = load_b_frag(pbase, 64);
        v16bf b1 = load_b_frag(pbase + 32, 64);
        v8f acc = zero8();
        acc = wmma_bf16(aS0, b0, acc);
        acc = wmma_bf16(aS1, b1, acc);
        int n = lane & 15, mb = (lane >> 4) * 8;
        int l = l0 + n;
        float dadd = Dmat[(size_t)h * TWO_L + l];
        for (int j = 0; j < 8; ++j) {
            int m = mb + j;
            int k = l + q0 + m;
            if (k < L_SZ) s_score[m][k] += acc[j] + dadd;
        }
    }
    // ---- Phase 3: boundary band (row q-1, high l2): k = l2 + r - 2L ----
    int tstart = (TWO_L - q0 - 14) >> 4;
    for (int t = tstart + wave; t < 128; t += 4) {
        int l20 = t * 16;
        const bf16_t* pbase = pebf + (size_t)l20 * 64;
        v16bf b0 = load_b_frag(pbase, 64);
        v16bf b1 = load_b_frag(pbase + 32, 64);
        v8f acc = zero8();
        acc = wmma_bf16(aB0, b0, acc);
        acc = wmma_bf16(aB1, b1, acc);
        int n = lane & 15, mb = (lane >> 4) * 8;
        int l2 = l20 + n;
        float dadd = Dmat[(size_t)h * TWO_L + l2];
        for (int j = 0; j < 8; ++j) {
            int m = mb + j;
            int r = q0 - 1 + m;       // source row = q-1
            int k = l2 + r - TWO_L;   // k = l2 + q - (2L+1)
            if (k >= 0 && k <= r - 1) s_score[m][k] += acc[j] + dadd;
        }
    }
    __syncthreads();

    // ---- Softmax over k (with mask), 8 threads per row ----
    {
        int m = tid >> 3, sub = tid & 7;
        int kb = sub * 128;
        float mx = -1e30f;
        for (int k = kb; k < kb + 128; ++k) {
            float s = s_score[m][k];
            if (mask[b * L_SZ + k] == 0) { s = -1e30f; s_score[m][k] = s; }
            mx = fmaxf(mx, s);
        }
        s_red[m][sub] = mx;
        __syncthreads();
        float rmx = -1e30f;
        for (int i = 0; i < 8; ++i) rmx = fmaxf(rmx, s_red[m][i]);
        float sm = 0.f;
        for (int k = kb; k < kb + 128; ++k) {
            float e = __expf(s_score[m][k] - rmx);
            s_score[m][k] = e;
            sm += e;
        }
        s_red[m][8 + sub] = sm;
        __syncthreads();
        float rsum = 0.f;
        for (int i = 0; i < 8; ++i) rsum += s_red[m][8 + i];
        float inv = 1.0f / rsum;
        for (int k = kb; k < kb + 128; ++k)
            s_p[m][k] = (bf16_t)(s_score[m][k] * inv);
    }
    __syncthreads();

    // ---- PV: out_tile(16 x 64) = P(16x1024) @ V(1024x64); wave w -> dh subtile ----
    {
        int dh0 = wave * 16;
        const bf16_t* vbase = vtbf + (((size_t)(b * H_SZ + h) * DH_SZ) + dh0) * L_SZ;
        v8f acc = zero8();
        for (int k0 = 0; k0 < L_SZ; k0 += 32) {
            v16bf a  = load_a_frag(&s_p[0][0] + k0, lane & 15, L_SZ);
            v16bf bb = load_b_frag(vbase + k0, L_SZ);
            acc = wmma_bf16(a, bb, acc);
        }
        int n = lane & 15, mb = (lane >> 4) * 8;
        for (int j = 0; j < 8; ++j) {
            int q = q0 + mb + j;
            out[((size_t)b * L_SZ + q) * D_SZ + h * 64 + dh0 + n] = acc[j];
        }
    }
}

extern "C" void kernel_launch(void* const* d_in, const int* in_sizes, int n_in,
                              void* d_out, int out_size, void* d_ws, size_t ws_size,
                              hipStream_t stream) {
    (void)in_sizes; (void)n_in; (void)out_size; (void)ws_size;
    const float* x    = (const float*)d_in[0];
    const int*   mask = (const int*)d_in[1];
    const float* Wqv  = (const float*)d_in[2];
    const float* rrb  = (const float*)d_in[3];
    const float* rwb  = (const float*)d_in[4];
    float* out = (float*)d_out;

    char* ws = (char*)d_ws;
    bf16_t* xbf   = (bf16_t*)(ws);                          // 4 MB
    bf16_t* wbf   = (bf16_t*)(ws + (4u  << 20));            // 4 MB
    bf16_t* qbf   = (bf16_t*)(ws + (8u  << 20));            // 4 MB
    bf16_t* qprbf = (bf16_t*)(ws + (12u << 20));            // 4 MB
    bf16_t* vtbf  = (bf16_t*)(ws + (16u << 20));            // 4 MB
    bf16_t* pebf  = (bf16_t*)(ws + (20u << 20));            // 256 KB
    float*  Dmat  = (float*) (ws + (20u << 20) + (256u << 10)); // 128 KB

    const int NX = B_SZ * L_SZ * D_SZ;   // 2M
    const int NW = 2 * D_SZ * D_SZ;      // 2M
    convert_kernel<<<(NX + 255) / 256, 256, 0, stream>>>(x, xbf, NX);
    convert_kernel<<<(NW + 255) / 256, 256, 0, stream>>>(Wqv, wbf, NW);
    pe_kernel<<<TWO_L / 256, 256, 0, stream>>>(rwb, pebf, Dmat);

    qv_gemm_kernel<<<dim3((B_SZ * L_SZ) / 16, (2 * D_SZ) / 256), 128, 0, stream>>>(
        xbf, wbf, rrb, qbf, qprbf, vtbf);

    attn_kernel<<<dim3(L_SZ / 16, H_SZ, B_SZ), 128, 0, stream>>>(
        xbf, qbf, qprbf, vtbf, pebf, Dmat, mask, out);
}
